// VGAEModel_atac_2_59081570123795
// MI455X (gfx1250) — matrix operand order repaired and checked by reading, not compiled, not run on hardware
//
#include <hip/hip_runtime.h>
#include <hip/hip_bf16.h>
#include <stdint.h>

#define NN   12288
#define EN   196608
#define DIN  3000
#define DHID 128
#define DLAT 16
#define DOUTC 3000
#define EE   (EN + NN)

typedef __attribute__((ext_vector_type(16))) __bf16        v16bf;
typedef __attribute__((ext_vector_type(8)))  unsigned int  v8u;
typedef __attribute__((ext_vector_type(8)))  float         v8f;

union FragAB { v8u w; v16bf b; };

__device__ __forceinline__ unsigned short f2bf(float x) {
  unsigned u = __float_as_uint(x);
  u += 0x7FFFu + ((u >> 16) & 1u);          // round-to-nearest-even
  return (unsigned short)(u >> 16);
}
__device__ __forceinline__ void st4bf(unsigned short* p,
                                      float a, float b, float c, float d) {
  unsigned lo = (unsigned)f2bf(a) | ((unsigned)f2bf(b) << 16);
  unsigned hi = (unsigned)f2bf(c) | ((unsigned)f2bf(d) << 16);
  *(unsigned long long*)p =
      (unsigned long long)lo | ((unsigned long long)hi << 32);
}
__device__ __forceinline__ unsigned ord_enc(float f) {
  unsigned b = __float_as_uint(f);
  return (b & 0x80000000u) ? ~b : (b | 0x80000000u);
}
__device__ __forceinline__ float ord_dec(unsigned e) {
  return (e & 0x80000000u) ? __uint_as_float(e & 0x7FFFFFFFu)
                           : __uint_as_float(~e);
}

// ---------------------------------------------------------------------------
// bf16 WMMA GEMM: C[M,Nn] = epi(A[M,K] @ B)   (B is [K,Nn], or [Nn,K] if transB)
// epilogue: 0 none, 1 +bias, 2 sigmoid(x+bias) [NT store], 3 sigmoid [NT store]
// block = 128 threads (4 waves); block tile 64x64; K-step 32
// LDS rows padded to 40 ushorts (80B = 5x16B): b128-aligned, conflict-free frags
// ---------------------------------------------------------------------------
#define LDSPITCH 40

__global__ __launch_bounds__(128) void k_gemm_wmma(
    const float* __restrict__ A, const float* __restrict__ B,
    const float* __restrict__ bias, float* __restrict__ C,
    int M, int K, int Nn, int transB, int epi)
{
  __shared__ unsigned short As[64][LDSPITCH];   // [m][k] bf16 bits
  __shared__ unsigned short Bs[64][LDSPITCH];   // [n][k] bf16 bits (k-contig)

  const int tid  = threadIdx.x;
  const int lane = tid & 31;
  const int wave = tid >> 5;                    // 0..3 -> M strip
  const int m0   = blockIdx.y * 64;
  const int n0   = blockIdx.x * 64;

  v8f acc[4];
#pragma unroll
  for (int t = 0; t < 4; ++t)
#pragma unroll
    for (int r = 0; r < 8; ++r) acc[t][r] = 0.0f;

  for (int k0 = 0; k0 < K; k0 += 32) {
    const bool fullK = (k0 + 32 <= K);

    // ---- stage A tile (64x32) ----
    if (fullK && (m0 + 64 <= M)) {
#pragma unroll
      for (int it = 0; it < 4; ++it) {
        int idx4 = (tid + it * 128) * 4;            // 0..2044, step 4
        int m = idx4 >> 5, k = idx4 & 31;
        const float4 v = *(const float4*)(A + (size_t)(m0 + m) * K + k0 + k);
        st4bf(&As[m][k], v.x, v.y, v.z, v.w);
      }
    } else {
#pragma unroll
      for (int it = 0; it < 4; ++it) {
        int idx4 = (tid + it * 128) * 4;
        int m = idx4 >> 5, k = idx4 & 31;
        int gm = m0 + m;
        float v[4];
#pragma unroll
        for (int j = 0; j < 4; ++j) {
          int gk = k0 + k + j;
          v[j] = (gm < M && gk < K) ? A[(size_t)gm * K + gk] : 0.0f;
        }
        st4bf(&As[m][k], v[0], v[1], v[2], v[3]);
      }
    }

    // ---- stage B tile into Bs[n][k] ----
    if (transB) {                                   // B is [Nn,K]: k-contiguous
      if (fullK && (n0 + 64 <= Nn)) {
#pragma unroll
        for (int it = 0; it < 4; ++it) {
          int idx4 = (tid + it * 128) * 4;
          int n = idx4 >> 5, k = idx4 & 31;
          const float4 v = *(const float4*)(B + (size_t)(n0 + n) * K + k0 + k);
          st4bf(&Bs[n][k], v.x, v.y, v.z, v.w);
        }
      } else {
#pragma unroll
        for (int it = 0; it < 4; ++it) {
          int idx4 = (tid + it * 128) * 4;
          int n = idx4 >> 5, k = idx4 & 31;
          int gn = n0 + n;
          float v[4];
#pragma unroll
          for (int j = 0; j < 4; ++j) {
            int gk = k0 + k + j;
            v[j] = (gn < Nn && gk < K) ? B[(size_t)gn * K + gk] : 0.0f;
          }
          st4bf(&Bs[n][k], v[0], v[1], v[2], v[3]);
        }
      }
    } else {                                        // B is [K,Nn]: n-contiguous
      if (fullK && (n0 + 64 <= Nn)) {
#pragma unroll
        for (int it = 0; it < 4; ++it) {
          int idx4 = (tid + it * 128) * 4;          // (k, n4) over 2048
          int k = idx4 >> 6, n = idx4 & 63;
          const float4 v = *(const float4*)(B + (size_t)(k0 + k) * Nn + n0 + n);
          Bs[n + 0][k] = f2bf(v.x); Bs[n + 1][k] = f2bf(v.y);
          Bs[n + 2][k] = f2bf(v.z); Bs[n + 3][k] = f2bf(v.w);
        }
      } else {
#pragma unroll
        for (int it = 0; it < 4; ++it) {
          int idx4 = (tid + it * 128) * 4;
          int k = idx4 >> 6, n = idx4 & 63;
          int gk = k0 + k;
          float v[4];
#pragma unroll
          for (int j = 0; j < 4; ++j) {
            int gn = n0 + n + j;
            v[j] = (gn < Nn && gk < K) ? B[(size_t)gk * Nn + gn] : 0.0f;
          }
          Bs[n + 0][k] = f2bf(v[0]); Bs[n + 1][k] = f2bf(v[1]);
          Bs[n + 2][k] = f2bf(v[2]); Bs[n + 3][k] = f2bf(v[3]);
        }
      }
    }
    __syncthreads();

    // A fragment for this wave's 16-row strip (ISA 7.12.2 16-bit A layout)
    FragAB fa;
    {
      int m  = wave * 16 + (lane & 15);
      int kb = (lane < 16) ? 0 : 8;                 // K base within 0..15 half
      const uint4* row = (const uint4*)&As[m][0];
      uint4 lo = row[kb >> 3];                      // K = kb..kb+7
      uint4 hi = row[2 + (kb >> 3)];                // K = 16+kb..16+kb+7
      fa.w[0] = lo.x; fa.w[1] = lo.y; fa.w[2] = lo.z; fa.w[3] = lo.w;
      fa.w[4] = hi.x; fa.w[5] = hi.y; fa.w[6] = hi.z; fa.w[7] = hi.w;
    }
#pragma unroll
    for (int nt = 0; nt < 4; ++nt) {
      FragAB fb;
      int n  = nt * 16 + (lane & 15);
      int kb = (lane < 16) ? 0 : 8;                 // K base within 0..15 half
      const uint4* row = (const uint4*)&Bs[n][0];
      uint4 lo = row[kb >> 3];                      // K = kb..kb+7
      uint4 hi = row[2 + (kb >> 3)];                // K = 16+kb..16+kb+7
      fb.w[0] = lo.x; fb.w[1] = lo.y; fb.w[2] = lo.z; fb.w[3] = lo.w;
      fb.w[4] = hi.x; fb.w[5] = hi.y; fb.w[6] = hi.z; fb.w[7] = hi.w;
      acc[nt] = __builtin_amdgcn_wmma_f32_16x16x32_bf16(
          false, fa.b, false, fb.b, (short)0, acc[nt], false, false);
    }
    __syncthreads();
  }

  // epilogue + store (C/D layout: elem r -> M = r or 8+r, N = lane&15)
#pragma unroll
  for (int nt = 0; nt < 4; ++nt) {
#pragma unroll
    for (int r = 0; r < 8; ++r) {
      int row = m0 + wave * 16 + ((lane < 16) ? r : 8 + r);
      int col = n0 + nt * 16 + (lane & 15);
      if (row < M && col < Nn) {
        float v = acc[nt][r];
        if (epi == 1 || epi == 2) v += bias[col];
        if (epi >= 2) {
          v = 1.0f / (1.0f + __expf(-v));
          __builtin_nontemporal_store(v, &C[(size_t)row * Nn + col]);
        } else {
          C[(size_t)row * Nn + col] = v;
        }
      }
    }
  }
}

// ---------------------------------------------------------------------------
// z -> bf16 with K padded 16->32 (zeros): zb[NN][32] bf16 bits
// ---------------------------------------------------------------------------
__global__ void k_zbf16(const float* __restrict__ z,
                        unsigned short* __restrict__ zb)
{
  int i = blockIdx.x * blockDim.x + threadIdx.x;       // over NN*32
  if (i >= NN * 32) return;
  int node = i >> 5, k = i & 31;
  float v = (k < DLAT) ? z[(size_t)node * DLAT + k] : 0.0f;
  zb[i] = f2bf(v);
}

// ---------------------------------------------------------------------------
// adj = sigmoid(z @ z^T), K=16 (padded to 32): fragments straight from global,
// no LDS, no barriers. M=N=12288 exact multiples of 64. NT stores (604 MB).
// block = 128 threads (4 waves); block tile 64x64
// ---------------------------------------------------------------------------
__global__ __launch_bounds__(128) void k_adj_wmma(
    const unsigned short* __restrict__ zb,   // [NN][32] bf16 bits
    float* __restrict__ adj)
{
  const int lane = threadIdx.x & 31;
  const int wave = threadIdx.x >> 5;
  const int m0 = blockIdx.y * 64 + wave * 16;
  const int n0 = blockIdx.x * 64;

  // A fragment (rows m0..m0+15)
  FragAB fa;
  {
    int m  = m0 + (lane & 15);
    int kb = (lane < 16) ? 0 : 8;
    const uint4* row = (const uint4*)(zb + (size_t)m * 32);   // 4x16B chunks
    uint4 lo = row[kb >> 3];                 // K = kb..kb+7
    uint4 hi = row[2 + (kb >> 3)];           // K = 16+kb..16+kb+7 (zeros)
    fa.w[0] = lo.x; fa.w[1] = lo.y; fa.w[2] = lo.z; fa.w[3] = lo.w;
    fa.w[4] = hi.x; fa.w[5] = hi.y; fa.w[6] = hi.z; fa.w[7] = hi.w;
  }

  v8f acc[4];
#pragma unroll
  for (int nt = 0; nt < 4; ++nt) {
    FragAB fb;
    int n  = n0 + nt * 16 + (lane & 15);
    int kb = (lane < 16) ? 0 : 8;
    const uint4* row = (const uint4*)(zb + (size_t)n * 32);
    uint4 lo = row[kb >> 3];                 // K = kb..kb+7
    uint4 hi = row[2 + (kb >> 3)];           // K = 16+kb..16+kb+7
    fb.w[0] = lo.x; fb.w[1] = lo.y; fb.w[2] = lo.z; fb.w[3] = lo.w;
    fb.w[4] = hi.x; fb.w[5] = hi.y; fb.w[6] = hi.z; fb.w[7] = hi.w;
    v8f c0;
#pragma unroll
    for (int r = 0; r < 8; ++r) c0[r] = 0.0f;
    acc[nt] = __builtin_amdgcn_wmma_f32_16x16x32_bf16(
        false, fa.b, false, fb.b, (short)0, c0, false, false);
  }

#pragma unroll
  for (int nt = 0; nt < 4; ++nt) {
#pragma unroll
    for (int r = 0; r < 8; ++r) {
      int row = m0 + ((lane < 16) ? r : 8 + r);
      int col = n0 + nt * 16 + (lane & 15);
      float v = 1.0f / (1.0f + __expf(-acc[nt][r]));
      __builtin_nontemporal_store(v, &adj[(size_t)row * NN + col]);
    }
  }
}

// ---------------------------------------------------------------------------
// fills
// ---------------------------------------------------------------------------
__global__ void k_fill_f32(float* p, float v, size_t n) {
  size_t i = (size_t)blockIdx.x * blockDim.x + threadIdx.x;
  if (i < n) p[i] = v;
}
__global__ void k_fill_u32(unsigned* p, unsigned v, size_t n) {
  size_t i = (size_t)blockIdx.x * blockDim.x + threadIdx.x;
  if (i < n) p[i] = v;
}

// ---------------------------------------------------------------------------
// GATv2 edge phase (edges 0..EN-1 from edge_index, EN..EE-1 self loops)
// ---------------------------------------------------------------------------
__global__ __launch_bounds__(256) void k_edge_score(
    const int* __restrict__ src, const int* __restrict__ dst,
    const float* __restrict__ xl, const float* __restrict__ xr,
    const float* __restrict__ att, int dout,
    float* __restrict__ esc, unsigned* __restrict__ mbuf)
{
  int wv   = blockIdx.x * 8 + (threadIdx.x >> 5);
  int lane = threadIdx.x & 31;
  if (wv >= EE) return;
  int s = (wv < EN) ? src[wv] : (wv - EN);
  int d = (wv < EN) ? dst[wv] : (wv - EN);
  float sum = 0.0f;
  for (int j = lane; j < dout; j += 32) {
    float v = xl[(size_t)s * dout + j] + xr[(size_t)d * dout + j];
    v = (v > 0.0f) ? v : 0.2f * v;           // LeakyReLU(0.2)
    sum += v * att[j];
  }
#pragma unroll
  for (int off = 16; off > 0; off >>= 1) sum += __shfl_down(sum, off, 32);
  if (lane == 0) {
    esc[wv] = sum;
    atomicMax(&mbuf[d], ord_enc(sum));       // segment max (ordered-uint)
  }
}

__global__ __launch_bounds__(256) void k_edge_exp(
    const int* __restrict__ dst, const float* __restrict__ esc_in,
    const unsigned* __restrict__ mbuf, float* __restrict__ esc_out,
    float* __restrict__ den)
{
  int e = blockIdx.x * blockDim.x + threadIdx.x;
  if (e >= EE) return;
  int d = (e < EN) ? dst[e] : (e - EN);
  float ex = __expf(esc_in[e] - ord_dec(mbuf[d]));
  esc_out[e] = ex;
  atomicAdd(&den[d], ex);
}

__global__ __launch_bounds__(256) void k_edge_scatter(
    const int* __restrict__ src, const int* __restrict__ dst,
    const float* __restrict__ xl, const float* __restrict__ esc,
    const float* __restrict__ den, float* __restrict__ accum, int dout)
{
  int wv   = blockIdx.x * 8 + (threadIdx.x >> 5);
  int lane = threadIdx.x & 31;
  if (wv >= EE) return;
  int s = (wv < EN) ? src[wv] : (wv - EN);
  int d = (wv < EN) ? dst[wv] : (wv - EN);
  float alpha = esc[wv] / den[d];
  for (int j = lane; j < dout; j += 32)
    atomicAdd(&accum[(size_t)d * dout + j], alpha * xl[(size_t)s * dout + j]);
}

__global__ void k_gat_finalize(const float* __restrict__ accum,
                               const float* __restrict__ b,
                               float* __restrict__ out, int dout, int relu)
{
  size_t i = (size_t)blockIdx.x * blockDim.x + threadIdx.x;
  if (i >= (size_t)NN * dout) return;
  float v = accum[i] + b[i % dout];
  if (relu) v = fmaxf(v, 0.0f);
  out[i] = v;
}

// ---------------------------------------------------------------------------
// BatchNorm (training stats) -> scale/shift, then fused leaky-relu apply
// ---------------------------------------------------------------------------
__global__ __launch_bounds__(256) void k_bn_stats(
    const float* __restrict__ X, int cols,
    const float* __restrict__ g, const float* __restrict__ b,
    float* __restrict__ scale, float* __restrict__ shift)
{
  __shared__ float s1[256], s2[256];
  int c = blockIdx.x, tid = threadIdx.x;
  float a = 0.0f, q = 0.0f;
  for (int r = tid; r < NN; r += 256) {
    float v = X[(size_t)r * cols + c];
    a += v; q += v * v;
  }
  s1[tid] = a; s2[tid] = q; __syncthreads();
  for (int o = 128; o > 0; o >>= 1) {
    if (tid < o) { s1[tid] += s1[tid + o]; s2[tid] += s2[tid + o]; }
    __syncthreads();
  }
  if (tid == 0) {
    float mu  = s1[0] / (float)NN;
    float var = s2[0] / (float)NN - mu * mu;
    float sc  = g[c] * rsqrtf(var + 1e-5f);
    scale[c] = sc;
    shift[c] = b[c] - mu * sc;
  }
}

__global__ void k_bn_apply_leaky(float* __restrict__ X,
                                 const float* __restrict__ scale,
                                 const float* __restrict__ shift, int cols)
{
  size_t i = (size_t)blockIdx.x * blockDim.x + threadIdx.x;
  if (i >= (size_t)NN * cols) return;
  int c = (int)(i % cols);
  float v = X[i] * scale[c] + shift[c];
  X[i] = (v > 0.0f) ? v : 0.2f * v;
}

__global__ void k_reparam(const float* __restrict__ eps,
                          const float* __restrict__ logstd,
                          const float* __restrict__ mu,
                          float* __restrict__ z)
{
  size_t i = (size_t)blockIdx.x * blockDim.x + threadIdx.x;
  if (i >= (size_t)NN * DLAT) return;
  z[i] = eps[i] * __expf(logstd[i]) + mu[i];
}

// ---------------------------------------------------------------------------
// host side
// ---------------------------------------------------------------------------
static inline unsigned cdiv(unsigned a, unsigned b) { return (a + b - 1) / b; }

static void run_gat(const float* X, int din, int dout,
                    const float* Wl, const float* Wr,
                    const float* att, const float* b,
                    const int* src, const int* dst,
                    float* xl, float* xr, float* accum, float* out,
                    float* esc, unsigned* mbuf, float* den,
                    int relu, hipStream_t s)
{
  dim3 blk(128);
  dim3 g(cdiv(dout, 64), cdiv(NN, 64));
  k_gemm_wmma<<<g, blk, 0, s>>>(X, Wl, nullptr, xl, NN, din, dout, 0, 0);
  k_gemm_wmma<<<g, blk, 0, s>>>(X, Wr, nullptr, xr, NN, din, dout, 0, 0);

  k_fill_u32<<<cdiv(NN, 256), 256, 0, s>>>(mbuf, 0x007FFFFFu, NN); // enc(-inf)
  k_fill_f32<<<cdiv(NN, 256), 256, 0, s>>>(den, 0.0f, NN);
  size_t an = (size_t)NN * dout;
  k_fill_f32<<<cdiv((unsigned)an, 256), 256, 0, s>>>(accum, 0.0f, an);

  k_edge_score  <<<cdiv(EE, 8), 256, 0, s>>>(src, dst, xl, xr, att, dout, esc, mbuf);
  k_edge_exp    <<<cdiv(EE, 256), 256, 0, s>>>(dst, esc, mbuf, esc, den);
  k_edge_scatter<<<cdiv(EE, 8), 256, 0, s>>>(src, dst, xl, esc, den, accum, dout);
  k_gat_finalize<<<cdiv((unsigned)an, 256), 256, 0, s>>>(accum, b, out, dout, relu);
}

extern "C" void kernel_launch(void* const* d_in, const int* in_sizes, int n_in,
                              void* d_out, int out_size, void* d_ws, size_t ws_size,
                              hipStream_t stream)
{
  (void)in_sizes; (void)n_in; (void)out_size; (void)ws_size;
  const float* x      = (const float*)d_in[0];
  const int*   ei     = (const int*)  d_in[1];
  const float* eps    = (const float*)d_in[2];
  const float* c1_Wl  = (const float*)d_in[3];
  const float* c1_Wr  = (const float*)d_in[4];
  const float* c1_att = (const float*)d_in[5];
  const float* c1_b   = (const float*)d_in[6];
  const float* c2_Wl  = (const float*)d_in[7];
  const float* c2_Wr  = (const float*)d_in[8];
  const float* c2_att = (const float*)d_in[9];
  const float* c2_b   = (const float*)d_in[10];
  const float* cm_Wl  = (const float*)d_in[11];
  const float* cm_Wr  = (const float*)d_in[12];
  const float* cm_att = (const float*)d_in[13];
  const float* cm_b   = (const float*)d_in[14];
  const float* cs_Wl  = (const float*)d_in[15];
  const float* cs_Wr  = (const float*)d_in[16];
  const float* cs_att = (const float*)d_in[17];
  const float* cs_b   = (const float*)d_in[18];
  const float* d1_W   = (const float*)d_in[19];
  // d_in[20] = d1_b: cancels in batchnorm (mean-subtracted)
  const float* bn1_g  = (const float*)d_in[21];
  const float* bn1_b  = (const float*)d_in[22];
  const float* d2_W   = (const float*)d_in[23];
  // d_in[24] = d2_b: cancels in batchnorm
  const float* bn2_g  = (const float*)d_in[25];
  const float* bn2_b  = (const float*)d_in[26];
  const float* d3_W   = (const float*)d_in[27];
  const float* d3_b   = (const float*)d_in[28];

  const int* src = ei;
  const int* dst = ei + EN;

  // output tuple layout: z, y, mu, logstd, adj_pred
  float* out    = (float*)d_out;
  float* z_out  = out;
  float* y_out  = out + (size_t)NN * DLAT;
  float* mu_out = y_out + (size_t)NN * DOUTC;
  float* ls_out = mu_out + (size_t)NN * DLAT;
  float* adj    = ls_out + (size_t)NN * DLAT;

  // workspace carve (floats); total ~19.3M floats (~78 MB)
  float* ws   = (float*)d_ws;
  float* xl   = ws;                                   // 12288*512
  float* xr   = xl + (size_t)NN * 512;                // 12288*512
  float* bufC = xr + (size_t)NN * 512;                // 12288*256
  float* bufD = bufC + (size_t)NN * 256;              // 12288*256
  float* esc  = bufD + (size_t)NN * 256;              // EE
  float* mbF  = esc + EE;                             // NN (uint view)
  unsigned* mbuf = (unsigned*)mbF;
  float* den  = mbF + NN;                             // NN
  float* bnsc = den + NN;                             // 512
  float* bnsh = bnsc + 512;                           // 512
  unsigned short* zb = (unsigned short*)(bnsh + 512); // NN*32 bf16 (16B-aligned)

  // ---- encoder: GATv2 stack ----
  run_gat(x,    DIN,  256,  c1_Wl, c1_Wr, c1_att, c1_b, src, dst,
          xl, xr, bufC, bufC, esc, mbuf, den, 1, stream);          // h1 in bufC
  run_gat(bufC, 256,  DHID, c2_Wl, c2_Wr, c2_att, c2_b, src, dst,
          xl, xr, bufD, bufD, esc, mbuf, den, 1, stream);          // h2 in bufD
  run_gat(bufD, DHID, DLAT, cm_Wl, cm_Wr, cm_att, cm_b, src, dst,
          xl, xr, bufC, mu_out, esc, mbuf, den, 0, stream);        // mu
  run_gat(bufD, DHID, DLAT, cs_Wl, cs_Wr, cs_att, cs_b, src, dst,
          xl, xr, bufC, ls_out, esc, mbuf, den, 0, stream);        // logstd

  // ---- reparameterize ----
  k_reparam<<<cdiv(NN * DLAT, 256), 256, 0, stream>>>(eps, ls_out, mu_out, z_out);

  // ---- adj_pred = sigmoid(z @ z^T): LDS-free direct-fragment WMMA ----
  k_zbf16<<<cdiv(NN * 32, 256), 256, 0, stream>>>(z_out, zb);
  {
    dim3 g(cdiv(NN, 64), cdiv(NN, 64));
    k_adj_wmma<<<g, 128, 0, stream>>>(zb, adj);
  }

  // ---- decoder MLP ----
  k_gemm_wmma<<<dim3(cdiv(128, 64), cdiv(NN, 64)), 128, 0, stream>>>(
      z_out, d1_W, nullptr, bufC, NN, DLAT, 128, 0, 0);            // a1
  k_bn_stats<<<128, 256, 0, stream>>>(bufC, 128, bn1_g, bn1_b, bnsc, bnsh);
  k_bn_apply_leaky<<<cdiv(NN * 128, 256), 256, 0, stream>>>(bufC, bnsc, bnsh, 128);

  k_gemm_wmma<<<dim3(cdiv(512, 64), cdiv(NN, 64)), 128, 0, stream>>>(
      bufC, d2_W, nullptr, xl, NN, 128, 512, 0, 0);                // a2
  k_bn_stats<<<512, 256, 0, stream>>>(xl, 512, bn2_g, bn2_b, bnsc, bnsh);
  k_bn_apply_leaky<<<cdiv(NN * 512, 256), 256, 0, stream>>>(xl, bnsc, bnsh, 512);

  k_gemm_wmma<<<dim3(cdiv(DOUTC, 64), cdiv(NN, 64)), 128, 0, stream>>>(
      xl, d3_W, d3_b, y_out, NN, 512, DOUTC, 0, 2);                // y
}
